// autoencoderLSTM_7773890806125
// MI455X (gfx1250) — matrix operand order repaired and checked by reading, not compiled, not run on hardware
//
#include <hip/hip_runtime.h>
#include <hip/hip_bf16.h>
#include <math.h>

// ---------------------------------------------------------------------------
// Types for CDNA5 WMMA
// ---------------------------------------------------------------------------
typedef __attribute__((ext_vector_type(16))) __bf16 v16bf;
typedef __attribute__((ext_vector_type(8)))  float  v8f;

union FragU { v16bf v; uint4 q[2]; };

// Problem dims
#define TT 64
#define BB 32
#define FF 8192
#define HH 512
#define MM (TT*BB)        // 2048 tokens

// LDS tiling for the big GEMMs
#define BLK_M 128
#define BLK_N 256
#define KB    64
#define LDP   72          // padded LDS row stride (elements): conflict-free

// ---------------------------------------------------------------------------
// CDNA5 async copy: global -> LDS, 16B per lane, tracked by ASYNCcnt
// ---------------------------------------------------------------------------
__device__ __forceinline__ void async_ld16(unsigned ldsAddr, const void* gAddr) {
  asm volatile("global_load_async_to_lds_b128 %0, %1, off"
               :: "v"(ldsAddr), "v"(gAddr) : "memory");
}
__device__ __forceinline__ void wait_async0() {
  asm volatile("s_wait_asynccnt 0x0" ::: "memory");
}

// ---------------------------------------------------------------------------
// fp32 -> bf16 (RNE) as raw u16 bits
// ---------------------------------------------------------------------------
__device__ __forceinline__ unsigned short f2bf(float x) {
  unsigned u = __float_as_uint(x);
  u += 0x7FFFu + ((u >> 16) & 1u);
  return (unsigned short)(u >> 16);
}

// ---------------------------------------------------------------------------
// Fragment loaders from row-major bf16 (global): used by the LSTM step GEMM
// A 16x32 bf16: lanes 0-15 -> M=lane, K = {k0..k0+7, k0+16..k0+23}
//               lanes 16-31 -> M=lane-16, K = {k0+8..k0+15, k0+24..k0+31}
// ---------------------------------------------------------------------------
__device__ __forceinline__ v16bf load_fragA(const unsigned short* __restrict__ base,
                                            int ld, int row0, int k0, int lane) {
  int r  = row0 + (lane & 15);
  int kb = k0 + ((lane & 16) ? 8 : 0);
  const unsigned short* p = base + (size_t)r * ld + kb;
  FragU f;
  f.q[0] = *(const uint4*)(p);
  f.q[1] = *(const uint4*)(p + 16);
  return f.v;
}

// B 32x16 bf16 from transposed weights Bt (N x K row-major):
// lanes 0-15 -> N=lane, K = {k0..k0+7, k0+16..k0+23} ; lanes 16-31 -> N=lane-16, K = {k0+8..k0+15, k0+24..k0+31}
__device__ __forceinline__ v16bf load_fragB(const unsigned short* __restrict__ baseT,
                                            int ld, int n0, int k0, int lane) {
  int r  = n0 + (lane & 15);
  int kb = k0 + ((lane & 16) ? 8 : 0);
  const unsigned short* p = baseT + (size_t)r * ld + kb;
  FragU f;
  f.q[0] = *(const uint4*)(p);
  f.q[1] = *(const uint4*)(p + 16);
  return f.v;
}

// Same fragment shapes, sourced from the padded LDS tiles
__device__ __forceinline__ v16bf lds_fragA(const unsigned short* s, int r0, int kk, int lane) {
  int r  = r0 + (lane & 15);
  int kb = kk + ((lane & 16) ? 8 : 0);
  const unsigned short* p = s + r * LDP + kb;
  FragU f;
  f.q[0] = *(const uint4*)(p);
  f.q[1] = *(const uint4*)(p + 16);
  return f.v;
}
__device__ __forceinline__ v16bf lds_fragB(const unsigned short* s, int r0, int kk, int lane) {
  int r  = r0 + (lane & 15);
  int kb = kk + ((lane & 16) ? 8 : 0);
  const unsigned short* p = s + r * LDP + kb;
  FragU f;
  f.q[0] = *(const uint4*)(p);
  f.q[1] = *(const uint4*)(p + 16);
  return f.v;
}

// ---------------------------------------------------------------------------
// Elementwise fp32 -> bf16 (vectorized x4)
// ---------------------------------------------------------------------------
__global__ __launch_bounds__(256) void cvt_bf16_v4(const float* __restrict__ in,
                                                   unsigned short* __restrict__ out,
                                                   int n4) {
  int i = blockIdx.x * blockDim.x + threadIdx.x;
  if (i < n4) {
    float4 v = ((const float4*)in)[i];
    unsigned lo = (unsigned)f2bf(v.x) | ((unsigned)f2bf(v.y) << 16);
    unsigned hi = (unsigned)f2bf(v.z) | ((unsigned)f2bf(v.w) << 16);
    ((uint2*)out)[i] = make_uint2(lo, hi);
  }
}

// ---------------------------------------------------------------------------
// Tiled transpose + convert: in (R x C fp32) -> out (C x R bf16)
// ---------------------------------------------------------------------------
__global__ __launch_bounds__(256) void transpose_cvt(const float* __restrict__ in,
                                                     unsigned short* __restrict__ out,
                                                     int R, int C) {
  __shared__ unsigned short tile[32][33];
  int c0 = blockIdx.x * 32, r0 = blockIdx.y * 32;
  int tx = threadIdx.x & 31, ty = threadIdx.x >> 5;   // ty: 0..7
#pragma unroll
  for (int i = 0; i < 32; i += 8)
    tile[ty + i][tx] = f2bf(in[(size_t)(r0 + ty + i) * C + (c0 + tx)]);
  __syncthreads();
#pragma unroll
  for (int i = 0; i < 32; i += 8)
    out[(size_t)(c0 + ty + i) * R + (r0 + tx)] = tile[tx][ty + i];
}

// ---------------------------------------------------------------------------
// Async-pipelined LDS GEMM: C(M,N) = act(A(M,K) @ Bt(N,K)^T + bias)
// Block = 256 threads = 8 waves (2x4) -> 128x256 tile per block.
// Double-buffered LDS tiles filled by global_load_async_to_lds_b128:
//   wait_asynccnt 0 ; barrier ; issue fills for k+64 ; 32x WMMA on tile k.
// One barrier per K-step; fills never touch VGPRs.
// ---------------------------------------------------------------------------
template<bool RELU, bool OUT_BF16>
__global__ __launch_bounds__(256)
void gemm_lds(const unsigned short* __restrict__ A,
              const unsigned short* __restrict__ Bt,
              const float* __restrict__ bias,
              float* __restrict__ Cf,
              unsigned short* __restrict__ Cb,
              int N, int K, int tilesN) {
  __shared__ unsigned short sA[2][BLK_M * LDP];
  __shared__ unsigned short sB[2][BLK_N * LDP];

  const int tid  = threadIdx.x;
  const int lane = tid & 31;
  const int wid  = tid >> 5;
  const int wm   = (wid >> 2) * 64;             // 0 or 64
  const int wn   = (wid & 3) * 64;              // 0,64,128,192
  const int bm   = (blockIdx.x / tilesN) * BLK_M;
  const int bn   = (blockIdx.x % tilesN) * BLK_N;

  // Fill map: thread covers one 64B half-row (32 elems) per 128-row panel
  const int row2 = tid >> 1;                    // 0..127
  const int half = (tid & 1) * 32;              // elem offset 0/32

  const unsigned short* gA  = A  + (size_t)(bm + row2) * K + half;
  const unsigned short* gB0 = Bt + (size_t)(bn + row2) * K + half;
  const unsigned short* gB1 = Bt + (size_t)(bn + 128 + row2) * K + half;

  unsigned ldsA[2], ldsB0[2], ldsB1[2];
#pragma unroll
  for (int b = 0; b < 2; ++b) {
    ldsA[b]  = (unsigned)(size_t)(&sA[b][row2 * LDP + half]);
    ldsB0[b] = (unsigned)(size_t)(&sB[b][row2 * LDP + half]);
    ldsB1[b] = (unsigned)(size_t)(&sB[b][(128 + row2) * LDP + half]);
  }

  // Prologue: tile k=0 -> buffer 0
#pragma unroll
  for (int j = 0; j < 4; ++j) {
    async_ld16(ldsA[0]  + j * 16, gA  + j * 8);
    async_ld16(ldsB0[0] + j * 16, gB0 + j * 8);
    async_ld16(ldsB1[0] + j * 16, gB1 + j * 8);
  }

  v8f acc[4][4] = {};
  int buf = 0;

  for (int k = 0; k < K; k += KB, buf ^= 1) {
    wait_async0();                              // own fills landed
    __syncthreads();                            // everyone's fills visible

    if (k + KB < K) {                           // fill other buffer for k+64
      const unsigned short* pA  = gA  + k + KB;
      const unsigned short* pB0 = gB0 + k + KB;
      const unsigned short* pB1 = gB1 + k + KB;
      const int nb = buf ^ 1;
#pragma unroll
      for (int j = 0; j < 4; ++j) {
        async_ld16(ldsA[nb]  + j * 16, pA  + j * 8);
        async_ld16(ldsB0[nb] + j * 16, pB0 + j * 8);
        async_ld16(ldsB1[nb] + j * 16, pB1 + j * 8);
      }
    }

    const unsigned short* cA = sA[buf];
    const unsigned short* cB = sB[buf];
#pragma unroll
    for (int kk = 0; kk < KB; kk += 32) {
      v16bf af[4], bf[4];
#pragma unroll
      for (int i = 0; i < 4; ++i) af[i] = lds_fragA(cA, wm + 16 * i, kk, lane);
#pragma unroll
      for (int j = 0; j < 4; ++j) bf[j] = lds_fragB(cB, wn + 16 * j, kk, lane);
#pragma unroll
      for (int i = 0; i < 4; ++i)
#pragma unroll
        for (int j = 0; j < 4; ++j)
          acc[i][j] = __builtin_amdgcn_wmma_f32_16x16x32_bf16(
              false, af[i], false, bf[j], (short)0, acc[i][j], false, false);
    }
  }

  // Epilogue: bias + activation + store
  const int m0 = bm + wm, n0 = bn + wn;
  const int lr = (lane & 16) ? 8 : 0;
  const int lc = lane & 15;
#pragma unroll
  for (int j = 0; j < 4; ++j) {
    const int col = n0 + 16 * j + lc;
    const float bv = bias[col];
#pragma unroll
    for (int i = 0; i < 4; ++i) {
#pragma unroll
      for (int v = 0; v < 8; ++v) {
        const int row = m0 + 16 * i + lr + v;
        float x = acc[i][j][v] + bv;
        if (RELU) x = fmaxf(x, 0.0f);
        if (OUT_BF16) Cb[(size_t)row * N + col] = f2bf(x);
        else          Cf[(size_t)row * N + col] = x;
      }
    }
  }
}

// ---------------------------------------------------------------------------
// LSTM step GEMM: Z(32,2048) = Xt(32,512)@Wx + H(32,512)@Wh
// One wave computes a 32x64 tile (2x4 fragments); 32 waves total (4 blocks).
// Weights are L2-resident (4 MB bf16) -> direct-global fragments are fine.
// ---------------------------------------------------------------------------
__global__ __launch_bounds__(256)
void lstm_gemm(const unsigned short* __restrict__ Xt,
               const unsigned short* __restrict__ Hb,
               const unsigned short* __restrict__ Wxt,
               const unsigned short* __restrict__ Wht,
               float* __restrict__ Z) {
  const int wave = blockIdx.x * 8 + (threadIdx.x >> 5);
  const int lane = threadIdx.x & 31;
  const int n0 = wave * 64;

  v8f acc[2][4] = {};

  for (int k = 0; k < HH; k += 32) {
    v16bf ax[2], bx[4];
#pragma unroll
    for (int i = 0; i < 2; ++i) ax[i] = load_fragA(Xt, HH, 16 * i, k, lane);
#pragma unroll
    for (int j = 0; j < 4; ++j) bx[j] = load_fragB(Wxt, HH, n0 + 16 * j, k, lane);
#pragma unroll
    for (int i = 0; i < 2; ++i)
#pragma unroll
      for (int j = 0; j < 4; ++j)
        acc[i][j] = __builtin_amdgcn_wmma_f32_16x16x32_bf16(
            false, ax[i], false, bx[j], (short)0, acc[i][j], false, false);

    v16bf ah[2], bh[4];
#pragma unroll
    for (int i = 0; i < 2; ++i) ah[i] = load_fragA(Hb, HH, 16 * i, k, lane);
#pragma unroll
    for (int j = 0; j < 4; ++j) bh[j] = load_fragB(Wht, HH, n0 + 16 * j, k, lane);
#pragma unroll
    for (int i = 0; i < 2; ++i)
#pragma unroll
      for (int j = 0; j < 4; ++j)
        acc[i][j] = __builtin_amdgcn_wmma_f32_16x16x32_bf16(
            false, ah[i], false, bh[j], (short)0, acc[i][j], false, false);
  }

  const int lr = (lane & 16) ? 8 : 0;
  const int lc = lane & 15;
#pragma unroll
  for (int i = 0; i < 2; ++i)
#pragma unroll
    for (int j = 0; j < 4; ++j)
#pragma unroll
      for (int v = 0; v < 8; ++v)
        Z[(size_t)(16 * i + lr + v) * (4 * HH) + (n0 + 16 * j + lc)] = acc[i][j][v];
}

// ---------------------------------------------------------------------------
// LSTM gates (Keras order [i,f,g,o]); writes c, h (fp32 + bf16) and Y[t]
// ---------------------------------------------------------------------------
__global__ __launch_bounds__(256)
void lstm_gates(const float* __restrict__ Z, const float* __restrict__ bgate,
                float* __restrict__ c, float* __restrict__ hf,
                unsigned short* __restrict__ hb, unsigned short* __restrict__ Y,
                int t) {
  int idx = blockIdx.x * blockDim.x + threadIdx.x;   // 0 .. B*H-1
  int bi = idx >> 9, j = idx & 511;
  const float* zr = Z + (size_t)bi * (4 * HH);
  float zi = zr[j]            + bgate[j];
  float zf = zr[HH + j]       + bgate[HH + j];
  float zg = zr[2 * HH + j]   + bgate[2 * HH + j];
  float zo = zr[3 * HH + j]   + bgate[3 * HH + j];
  float ig = 1.0f / (1.0f + __expf(-zi));
  float fg = 1.0f / (1.0f + __expf(-zf));
  float gg = tanhf(zg);
  float og = 1.0f / (1.0f + __expf(-zo));
  float cn = fg * c[idx] + ig * gg;
  float hn = og * tanhf(cn);
  c[idx]  = cn;
  hf[idx] = hn;
  unsigned short hbv = f2bf(hn);
  hb[idx] = hbv;
  Y[(size_t)(t * BB + bi) * HH + j] = hbv;
}

__global__ __launch_bounds__(256)
void init_state(const float* __restrict__ h0, const float* __restrict__ c0,
                float* __restrict__ hf, float* __restrict__ cf,
                unsigned short* __restrict__ hb) {
  int idx = blockIdx.x * blockDim.x + threadIdx.x;
  float h = h0[idx];
  hf[idx] = h;
  hb[idx] = f2bf(h);
  cf[idx] = c0[idx];
}

__global__ __launch_bounds__(256)
void copy_states(const float* __restrict__ hf, const float* __restrict__ cf,
                 float* __restrict__ outTail) {
  int idx = blockIdx.x * blockDim.x + threadIdx.x;   // 0 .. B*H-1
  outTail[idx]           = hf[idx];
  outTail[BB * HH + idx] = cf[idx];
}

// ---------------------------------------------------------------------------
// Host orchestration
// ---------------------------------------------------------------------------
extern "C" void kernel_launch(void* const* d_in, const int* in_sizes, int n_in,
                              void* d_out, int out_size, void* d_ws, size_t ws_size,
                              hipStream_t stream) {
  const float* inputs = (const float*)d_in[0];
  const float* h0     = (const float*)d_in[1];
  const float* c0     = (const float*)d_in[2];
  const float* We1    = (const float*)d_in[3];
  const float* be1    = (const float*)d_in[4];
  const float* We2    = (const float*)d_in[5];
  const float* be2    = (const float*)d_in[6];
  const float* Wx     = (const float*)d_in[7];
  const float* Wh     = (const float*)d_in[8];
  const float* bg     = (const float*)d_in[9];
  const float* Wd     = (const float*)d_in[10];
  const float* bd     = (const float*)d_in[11];

  char* ws = (char*)d_ws;
  size_t off = 0;
  auto carve = [&](size_t bytes) -> char* {
    char* p = ws + off;
    off += (bytes + 255) & ~(size_t)255;
    return p;
  };
  unsigned short* Abf = (unsigned short*)carve((size_t)MM * FF * 2);
  unsigned short* W1t = (unsigned short*)carve((size_t)2048 * FF * 2);
  unsigned short* X1  = (unsigned short*)carve((size_t)MM * 2048 * 2);
  unsigned short* W2t = (unsigned short*)carve((size_t)HH * 2048 * 2);
  unsigned short* X2  = (unsigned short*)carve((size_t)MM * HH * 2);
  unsigned short* Wxt = (unsigned short*)carve((size_t)4 * HH * HH * 2);
  unsigned short* Wht = (unsigned short*)carve((size_t)4 * HH * HH * 2);
  unsigned short* Wdt = (unsigned short*)carve((size_t)FF * HH * 2);
  unsigned short* Ybf = (unsigned short*)carve((size_t)MM * HH * 2);
  float*          Z   = (float*)carve((size_t)BB * 4 * HH * 4);
  float*          cst = (float*)carve((size_t)BB * HH * 4);
  float*          hf  = (float*)carve((size_t)BB * HH * 4);
  unsigned short* hb  = (unsigned short*)carve((size_t)BB * HH * 2);

  float* out = (float*)d_out;

  // 1) Convert activations to bf16
  {
    int n4 = MM * FF / 4;
    cvt_bf16_v4<<<(n4 + 255) / 256, 256, 0, stream>>>(inputs, Abf, n4);
  }
  // 2) Transpose+convert weights: (K x N fp32) -> (N x K bf16)
  transpose_cvt<<<dim3(2048 / 32, FF / 32), 256, 0, stream>>>(We1, W1t, FF, 2048);
  transpose_cvt<<<dim3(HH / 32, 2048 / 32), 256, 0, stream>>>(We2, W2t, 2048, HH);
  transpose_cvt<<<dim3(4 * HH / 32, HH / 32), 256, 0, stream>>>(Wx, Wxt, HH, 4 * HH);
  transpose_cvt<<<dim3(4 * HH / 32, HH / 32), 256, 0, stream>>>(Wh, Wht, HH, 4 * HH);
  transpose_cvt<<<dim3(FF / 32, HH / 32), 256, 0, stream>>>(Wd, Wdt, HH, FF);

  // 3) Encoder layer 1: X1 = relu(A @ We1 + be1)  [2048 x 2048], K=8192
  {
    int tilesN = 2048 / BLK_N;
    int blocks = (MM / BLK_M) * tilesN;
    gemm_lds<true, true><<<blocks, 256, 0, stream>>>(Abf, W1t, be1, nullptr, X1,
                                                     2048, FF, tilesN);
  }
  // 4) Encoder layer 2: X2 = relu(X1 @ We2 + be2)  [2048 x 512], K=2048
  {
    int tilesN = HH / BLK_N;
    int blocks = (MM / BLK_M) * tilesN;
    gemm_lds<true, true><<<blocks, 256, 0, stream>>>(X1, W2t, be2, nullptr, X2,
                                                     HH, 2048, tilesN);
  }
  // 5) LSTM over 64 timesteps
  init_state<<<(BB * HH) / 256, 256, 0, stream>>>(h0, c0, hf, cst, hb);
  for (int t = 0; t < TT; ++t) {
    lstm_gemm<<<4, 256, 0, stream>>>(X2 + (size_t)t * BB * HH, hb, Wxt, Wht, Z);
    lstm_gates<<<(BB * HH) / 256, 256, 0, stream>>>(Z, bg, cst, hf, hb, Ybf, t);
  }
  // 6) Decoder: out = Y @ Wd + bd  [2048 x 8192], K=512, fp32 out
  {
    int tilesN = FF / BLK_N;
    int blocks = (MM / BLK_M) * tilesN;
    gemm_lds<false, false><<<blocks, 256, 0, stream>>>(Ybf, Wdt, bd, out, nullptr,
                                                       FF, HH, tilesN);
  }
  // 7) Final states appended after the (T*B x 8192) output
  copy_states<<<(BB * HH) / 256, 256, 0, stream>>>(hf, cst, out + (size_t)MM * FF);

  (void)in_sizes; (void)n_in; (void)out_size; (void)ws_size;
}
